// Corr_18013092840196
// MI455X (gfx1250) — matrix operand order repaired and checked by reading, not compiled, not run on hardware
//
#include <hip/hip_runtime.h>
#include <hip/hip_bf16.h>

typedef _Float16 v16h __attribute__((ext_vector_type(16)));
typedef _Float16 v8h  __attribute__((ext_vector_type(8)));
typedef _Float16 h2   __attribute__((ext_vector_type(2)));
typedef float    v8f  __attribute__((ext_vector_type(8)));

#define BATCH 8
#define CCH   256
#define HW    40
#define P2    1600              // 40*40
#define LDSROW 40               // half stride per row (80B, keeps 16B alignment)

// ---------------------------------------------------------------------------
// K0: inverse L2 norms along channel for xp and xq.  idx < 2*8*1600
// ---------------------------------------------------------------------------
__global__ __launch_bounds__(256) void norms_kernel(const float* __restrict__ xp,
                                                    const float* __restrict__ xq,
                                                    float* __restrict__ invP,
                                                    float* __restrict__ invQ) {
  int idx = blockIdx.x * blockDim.x + threadIdx.x;
  if (idx >= 2 * BATCH * P2) return;
  int which = idx / (BATCH * P2);
  int r = idx % (BATCH * P2);          // b*1600 + p
  const float* src = which ? xq : xp;
  int b = r / P2, p = r % P2;
  const float* base = src + (size_t)b * CCH * P2 + p;
  float s = 0.f;
#pragma unroll 8
  for (int c = 0; c < CCH; ++c) {
    float v = base[(size_t)c * P2];
    s += v * v;
  }
  float inv = 1.0f / fmaxf(sqrtf(s), 1e-12f);
  (which ? invQ : invP)[r] = inv;
}

// ---------------------------------------------------------------------------
// K1: masked, scaled affinity GEMM  s[b][p][q] = a*mask*invP*invQ*sum_c xp*xq
//     64x64 tile, 4 waves (128 threads), f16 WMMA with f32 accumulate.
//     Double-buffered LDS, fully unrolled K loop (8 steps x 4 WMMA).
//     Staging: thread owns one column and 16 contiguous K values so
//     converted halves pack into ds_store_b32 (half2) stores.
//     Fragment layouts per CDNA5 ISA 7.12.2 (wave32).
// ---------------------------------------------------------------------------
__global__ __launch_bounds__(128) void gemm_kernel(const float* __restrict__ xp,
                                                   const float* __restrict__ xq,
                                                   const float* __restrict__ invP,
                                                   const float* __restrict__ invQ,
                                                   const float* __restrict__ alpha_p,
                                                   float* __restrict__ S) {
  __shared__ __align__(16) _Float16 As[2][64 * LDSROW];
  __shared__ __align__(16) _Float16 Bs[2][64 * LDSROW];

  const int b  = blockIdx.z;
  const int q0 = blockIdx.x * 64;
  const int p0 = blockIdx.y * 64;
  const int tid  = threadIdx.x;
  const int lane = tid & 31;
  const int wave = tid >> 5;           // 0..3 -> 16-row M strip

  const float* A  = xp + (size_t)b * CCH * P2;   // [c][p], p contiguous
  const float* Bg = xq + (size_t)b * CCH * P2;   // [c][q], q contiguous

  // Staging decomposition: each thread owns column `scol` and 16 contiguous
  // K values starting at `skk` within the 32-K slab.
  const int scol = tid & 63;           // 0..63
  const int skk  = (tid >> 6) * 16;    // 0 or 16
  const size_t gbase = (size_t)skk * P2;       // row offset within slab
  _Float16* const asBase0 = &As[0][scol * LDSROW + skk];
  _Float16* const bsBase0 = &Bs[0][scol * LDSROW + skk];
  _Float16* const asBase1 = &As[1][scol * LDSROW + skk];
  _Float16* const bsBase1 = &Bs[1][scol * LDSROW + skk];

  v8f acc[4] = {};                     // 4 N-subtiles of 16x16 f32

  // Stage K-step 0 into buffer 0 (packed half2 stores).
  {
    const float* ga = A  + gbase + p0 + scol;
    const float* gb = Bg + gbase + q0 + scol;
    float ar[16], br[16];
#pragma unroll
    for (int i = 0; i < 16; ++i) {
      ar[i] = ga[(size_t)i * P2];
      br[i] = gb[(size_t)i * P2];
    }
#pragma unroll
    for (int j = 0; j < 8; ++j) {
      *(h2*)(asBase0 + 2 * j) = h2{(_Float16)ar[2 * j], (_Float16)ar[2 * j + 1]};
      *(h2*)(bsBase0 + 2 * j) = h2{(_Float16)br[2 * j], (_Float16)br[2 * j + 1]};
    }
  }
  __syncthreads();

  // Fragment addresses (constant across K steps).
  const int mrow  = wave * 16 + (lane & 15);
  const int baseA = (lane >= 16) ? 8 : 0;
  const int baseB = (lane >= 16) ? 8 : 0;

#pragma unroll
  for (int step = 0; step < 8; ++step) {
    const int cur = step & 1;

    // Prefetch next K slab from global into registers (overlaps with WMMA).
    float ar[16], br[16];
    if (step < 7) {
      const size_t koff = gbase + (size_t)(step + 1) * 32 * P2;
      const float* ga = A  + koff + p0 + scol;
      const float* gb = Bg + koff + q0 + scol;
#pragma unroll
      for (int i = 0; i < 16; ++i) {
        ar[i] = ga[(size_t)i * P2];
        br[i] = gb[(size_t)i * P2];
      }
    }

    // A fragment: lane holds row M = wave*16 + lane%16.
    // lanes 0-15: K = {0..7, 16..23}; lanes 16-31: K = {8..15, 24..31}.
    v8h alo = *(const v8h*)(&As[cur][mrow * LDSROW + baseA]);
    v8h ahi = *(const v8h*)(&As[cur][mrow * LDSROW + baseA + 16]);
    v16h afrag;
#pragma unroll
    for (int i = 0; i < 8; ++i) { afrag[i] = alo[i]; afrag[i + 8] = ahi[i]; }

#pragma unroll
    for (int ns = 0; ns < 4; ++ns) {
      // B fragment: lane holds col N = ns*16 + lane%16.
      // lanes 0-15: K = {0..7, 16..23}; lanes 16-31: K = {8..15, 24..31}.
      int nrow = ns * 16 + (lane & 15);
      v8h blo = *(const v8h*)(&Bs[cur][nrow * LDSROW + baseB]);
      v8h bhi = *(const v8h*)(&Bs[cur][nrow * LDSROW + baseB + 16]);
      v16h bfrag;
#pragma unroll
      for (int i = 0; i < 8; ++i) { bfrag[i] = blo[i]; bfrag[i + 8] = bhi[i]; }

      acc[ns] = __builtin_amdgcn_wmma_f32_16x16x32_f16(
          false, afrag, false, bfrag, (short)0, acc[ns], false, false);
    }

    // Convert + store prefetched slab into the other buffer; one barrier/step.
    if (step < 7) {
      _Float16* as = (cur ? asBase0 : asBase1);
      _Float16* bs = (cur ? bsBase0 : bsBase1);
#pragma unroll
      for (int j = 0; j < 8; ++j) {
        *(h2*)(as + 2 * j) = h2{(_Float16)ar[2 * j], (_Float16)ar[2 * j + 1]};
        *(h2*)(bs + 2 * j) = h2{(_Float16)br[2 * j], (_Float16)br[2 * j + 1]};
      }
      __syncthreads();
    }
  }

  // Epilogue: D layout — vgpr r: M = r + (lane>=16 ? 8 : 0), N = lane%16.
  const float alpha = alpha_p[0];
  float* Sb = S + (size_t)b * P2 * P2;
  const int hi    = (lane >= 16) ? 8 : 0;
  const int nbase = lane & 15;
#pragma unroll
  for (int ns = 0; ns < 4; ++ns) {
    int q  = q0 + ns * 16 + nbase;
    int r2 = q / HW, c2 = q % HW;
    float iq = invQ[b * P2 + q];
#pragma unroll
    for (int r = 0; r < 8; ++r) {
      int p  = p0 + wave * 16 + r + hi;
      int r1 = p / HW, c1 = p % HW;
      float dr = (float)(r2 - r1);
      float dc = (float)(c2 - c1);
      // sr = sc = 40*0.05 = 2  ->  2*s^2 = 8
      float msk = 1.0f - __expf(-(dr * dr + dc * dc) * 0.125f);
      float ip  = invP[b * P2 + p];
      Sb[(size_t)p * P2 + q] = acc[ns][r] * ip * iq * alpha * msk;
    }
  }
}

// ---------------------------------------------------------------------------
// K2: per-row online softmax stats (max + 1/sumexp). One block per row.
// ---------------------------------------------------------------------------
__global__ __launch_bounds__(256) void row_stats_kernel(const float* __restrict__ S,
                                                        float* __restrict__ rowmax,
                                                        float* __restrict__ rowrcp) {
  int row = blockIdx.x;                  // b*1600 + p, 12800 rows
  const float* r = S + (size_t)row * P2;
  int tid = threadIdx.x;
  float m = -1e30f, s = 0.f;
  for (int q = tid; q < P2; q += 256) {
    float v  = r[q];
    float nm = fmaxf(m, v);
    s = s * __expf(m - nm) + __expf(v - nm);
    m = nm;
  }
  __shared__ float sm[256], ss[256];
  sm[tid] = m; ss[tid] = s;
  __syncthreads();
  for (int off = 128; off > 0; off >>= 1) {
    if (tid < off) {
      float m2 = sm[tid + off], s2 = ss[tid + off];
      float nm = fmaxf(sm[tid], m2);
      ss[tid] = ss[tid] * __expf(sm[tid] - nm) + s2 * __expf(m2 - nm);
      sm[tid] = nm;
    }
    __syncthreads();
  }
  if (tid == 0) { rowmax[row] = sm[0]; rowrcp[row] = 1.0f / ss[0]; }
}

// ---------------------------------------------------------------------------
// K3: per-column online softmax stats. Block = 32 q-columns x 8 p-lanes.
// ---------------------------------------------------------------------------
__global__ __launch_bounds__(256) void col_stats_kernel(const float* __restrict__ S,
                                                        float* __restrict__ colmax,
                                                        float* __restrict__ colrcp) {
  int b  = blockIdx.y;
  int q0 = blockIdx.x * 32;
  int tid = threadIdx.x;
  int qi = tid & 31, pi = tid >> 5;     // pi in 0..7
  const float* Sb = S + (size_t)b * P2 * P2;
  float m = -1e30f, s = 0.f;
  for (int p = pi; p < P2; p += 8) {
    float v  = Sb[(size_t)p * P2 + q0 + qi];
    float nm = fmaxf(m, v);
    s = s * __expf(m - nm) + __expf(v - nm);
    m = nm;
  }
  __shared__ float sm[8][32], ss[8][32];
  sm[pi][qi] = m; ss[pi][qi] = s;
  __syncthreads();
  if (pi == 0) {
    float M = sm[0][qi], Sv = ss[0][qi];
#pragma unroll
    for (int k = 1; k < 8; ++k) {
      float m2 = sm[k][qi], s2 = ss[k][qi];
      float nm = fmaxf(M, m2);
      Sv = Sv * __expf(M - nm) + s2 * __expf(m2 - nm);
      M = nm;
    }
    colmax[b * P2 + q0 + qi] = M;
    colrcp[b * P2 + q0 + qi] = 1.0f / Sv;
  }
}

// ---------------------------------------------------------------------------
// K4: x_c = softmax_row * softmax_col; in-place over S (-> x_soft_q) and
//     LDS-transposed write of x_soft_p. 32x32 tile per block.
// ---------------------------------------------------------------------------
__global__ __launch_bounds__(256) void finalize_kernel(const float* __restrict__ rowmax,
                                                       const float* __restrict__ rowrcp,
                                                       const float* __restrict__ colmax,
                                                       const float* __restrict__ colrcp,
                                                       float* __restrict__ S,      // in: s, out: x_soft_q
                                                       float* __restrict__ xsp) {  // out: x_soft_p
  __shared__ float tile[32][33];
  int b  = blockIdx.z;
  int p0 = blockIdx.y * 32, q0 = blockIdx.x * 32;
  float* Sb = S   + (size_t)b * P2 * P2;
  float* Pb = xsp + (size_t)b * P2 * P2;
  int tid = threadIdx.x;
  int qi = tid & 31, pr = tid >> 5;     // pr in 0..7
#pragma unroll
  for (int pass = 0; pass < 4; ++pass) {
    int pl = pr + pass * 8;
    int p = p0 + pl, q = q0 + qi;
    float v  = Sb[(size_t)p * P2 + q];
    float xc = __expf(v - rowmax[b * P2 + p]) * rowrcp[b * P2 + p] *
               __expf(v - colmax[b * P2 + q]) * colrcp[b * P2 + q];
    Sb[(size_t)p * P2 + q] = xc;        // x_soft_q (in-place, elementwise safe)
    tile[pl][qi] = xc;
  }
  __syncthreads();
#pragma unroll
  for (int pass = 0; pass < 4; ++pass) {
    int ql = pr + pass * 8;             // local q row of transposed tile
    Pb[(size_t)(q0 + ql) * P2 + p0 + qi] = tile[qi][ql];
  }
}

// ---------------------------------------------------------------------------
// K5: 8x8 block max pooling. valp from x_soft_q rows, valq from x_soft_p rows.
//     One thread per pooled output (2 * 8*25*1600 = 640000 threads).
// ---------------------------------------------------------------------------
__global__ __launch_bounds__(256) void blockmax_kernel(const float* __restrict__ xsq,
                                                       const float* __restrict__ xsp,
                                                       float* __restrict__ valp,
                                                       float* __restrict__ valq) {
  int idx = blockIdx.x * blockDim.x + threadIdx.x;
  if (idx >= 2 * BATCH * 25 * P2) return;
  bool second = idx >= BATCH * 25 * P2;
  int o = second ? idx - BATCH * 25 * P2 : idx;
  const float* src = second ? xsp : xsq;
  float* dst = second ? valq : valp;
  int b   = o / (25 * P2);
  int k   = (o / P2) % 25;
  int pix = o % P2;                     // flat (r,c) of the kept axis
  int br = k / 5, bc = k % 5;           // 8x8 block of the pooled axis
  const float* rowp = src + ((size_t)(b * P2 + pix)) * P2;
  float m = 0.f;                        // x_c > 0 always
#pragma unroll
  for (int r = 0; r < 8; ++r) {
    const float* rr = rowp + (br * 8 + r) * HW + bc * 8;
#pragma unroll
    for (int c = 0; c < 8; ++c) m = fmaxf(m, rr[c]);
  }
  dst[o] = m;
}

// ---------------------------------------------------------------------------
extern "C" void kernel_launch(void* const* d_in, const int* in_sizes, int n_in,
                              void* d_out, int out_size, void* d_ws, size_t ws_size,
                              hipStream_t stream) {
  const float* xp    = (const float*)d_in[0];
  const float* xq    = (const float*)d_in[1];
  const float* alpha = (const float*)d_in[2];

  float* out  = (float*)d_out;
  float* valp = out;                                   //  8*25*40*40 = 320000
  float* valq = out + 320000;                          //  320000
  float* xsp  = out + 640000;                          //  8*1600*1600
  float* xsq  = out + 640000 + BATCH * P2 * P2;        //  8*1600*1600 (also hosts s)

  float* ws     = (float*)d_ws;                        // only ~300 KB of stats
  float* invP   = ws;
  float* invQ   = ws + 1 * BATCH * P2;
  float* rowmax = ws + 2 * BATCH * P2;
  float* rowrcp = ws + 3 * BATCH * P2;
  float* colmax = ws + 4 * BATCH * P2;
  float* colrcp = ws + 5 * BATCH * P2;

  norms_kernel<<<(2 * BATCH * P2 + 255) / 256, 256, 0, stream>>>(xp, xq, invP, invQ);
  gemm_kernel<<<dim3(P2 / 64, P2 / 64, BATCH), 128, 0, stream>>>(xp, xq, invP, invQ, alpha, xsq);
  row_stats_kernel<<<BATCH * P2, 256, 0, stream>>>(xsq, rowmax, rowrcp);
  col_stats_kernel<<<dim3(P2 / 32, BATCH), 256, 0, stream>>>(xsq, colmax, colrcp);
  finalize_kernel<<<dim3(P2 / 32, P2 / 32, BATCH), 256, 0, stream>>>(rowmax, rowrcp, colmax, colrcp, xsq, xsp);
  blockmax_kernel<<<(2 * BATCH * 25 * P2 + 255) / 256, 256, 0, stream>>>(xsq, xsp, valp, valq);
}
